// SwitchGate_61478161875325
// MI455X (gfx1250) — matrix operand order repaired and checked by reading, not compiled, not run on hardware
//
#include <hip/hip_runtime.h>
#include <hip/hip_bf16.h>
#include <math.h>

// ---------------------------------------------------------------------------
// SwitchGate on gfx1250 (MI455X): TDM-staged bf16 WMMA gating GEMM + fused
// softmax, then mask/denominator/output/loss kernels. wave32 / WMMA path.
// ---------------------------------------------------------------------------

#define B_DIM   2048
#define N_SEQ   128
#define D_DIM   512
#define E_DIM   64
#define M_TOT   (B_DIM * N_SEQ)          // 262144 token rows
#define CAP     2048.0f                  // capacity = CAPACITY_FACTOR * B
#define EPS_G   1e-6f
#define EPS_CV  1e-10f

typedef __attribute__((ext_vector_type(16))) __bf16       v16bf;
typedef __attribute__((ext_vector_type(8)))  float        v8f;
typedef __attribute__((ext_vector_type(4)))  unsigned int v4ui;
typedef __attribute__((ext_vector_type(8)))  int          v8i;
typedef __attribute__((ext_vector_type(4)))  int          v4i;

// ---------------------------------------------------------------------------
// Kernel 0: one-time fp32 -> bf16 conversion of w_gate [E,D] into workspace.
// ---------------------------------------------------------------------------
__global__ __launch_bounds__(256)
void cvt_w_kernel(const float* __restrict__ w, __bf16* __restrict__ wbf)
{
    const int i = blockIdx.x * 256 + threadIdx.x;   // grid covers E*D = 32768
    wbf[i] = (__bf16)w[i];
}

// ---------------------------------------------------------------------------
// Kernel 1: logits = x @ w^T (+b) via v_wmma_f32_16x16x32_bf16, fused softmax.
// w (bf16, 64KB) is DMA'd into LDS once per block by the Tensor Data Mover;
// the logit staging buffer overlays the same LDS after the K loop.
// Block = 256 threads (8 waves). Wave -> 16 rows x 64 experts; block -> 128
// rows. Grid = M_TOT/128 = 2048 blocks.
// ---------------------------------------------------------------------------
__global__ __launch_bounds__(256)
void gate_gemm_softmax(const float*  __restrict__ x,
                       const __bf16* __restrict__ wbf,
                       const float*  __restrict__ bg,
                       float* __restrict__ p0,
                       int*   __restrict__ amax)
{
    // Phase A: 64KB bf16 copy of w_gate. Phase B (after barrier): logits.
    __shared__ __align__(32) unsigned char smem[E_DIM * D_DIM * 2];  // 64 KB
    __bf16* wlds = (__bf16*)smem;
    float (*slog)[E_DIM + 1] = (float (*)[E_DIM + 1])smem;

    const int lane = threadIdx.x & 31;
    const int wid  = threadIdx.x >> 5;       // 0..7
    const int r    = lane & 15;              // row-in-tile / col-in-tile
    const int h    = lane >> 4;              // lane half (K-group select)
    const size_t row0 = (size_t)blockIdx.x * 128;

    // ---- TDM: DMA the full 64x512 bf16 w matrix into LDS (one wave issues).
    if (threadIdx.x < 32) {
        const unsigned long long gaddr = (unsigned long long)(size_t)wbf;
        const unsigned int ldsoff = (unsigned int)(size_t)wlds;  // addr[31:0]
        // D# group 0: count=1 | lds_addr | global_addr[56:0] | type=2
        v4ui g0 = { 1u,
                    ldsoff,
                    (unsigned int)(gaddr & 0xFFFFFFFFull),
                    (unsigned int)((gaddr >> 32) & 0x01FFFFFFull) | 0x80000000u };
        // D# group 1: data_size=2B; tensor 512x64; tile 512x64; stride0=512
        v8i g1 = { 0x00010000,                 // wg_mask=0, data_size=1 (2B)
                   (int)(D_DIM << 16),          // tensor_dim0[15:0]=512
                   (int)(E_DIM << 16),          // tensor_dim0[31:16]=0, dim1=64
                   (int)(D_DIM << 16),          // tensor_dim1 hi=0, tile_dim0=512
                   E_DIM,                       // tile_dim1=64, tile_dim2=0
                   D_DIM,                       // tensor_dim0_stride=512
                   0, 0 };                      // stride0 hi, stride1=0
        v4i g2 = { 0, 0, 0, 0 };               // 2-D tile: groups 2/3 unused
        v4i g3 = { 0, 0, 0, 0 };
        v8i g4 = { 0, 0, 0, 0, 0, 0, 0, 0 };   // unused trailing group
        __builtin_amdgcn_tensor_load_to_lds(g0, g1, g2, g3, g4, /*cpol=*/0);
        __builtin_amdgcn_s_wait_tensorcnt(0);
    }
    __syncthreads();

    const float* xr = x + (row0 + (size_t)wid * 16 + r) * D_DIM;

    v8f acc[4];
#pragma unroll
    for (int ct = 0; ct < 4; ++ct) acc[ct] = (v8f){0.f,0.f,0.f,0.f,0.f,0.f,0.f,0.f};

    for (int kt = 0; kt < D_DIM / 32; ++kt) {
        // ---- A fragment: 16x32 bf16. Lane (r,h) holds K = h*16 .. h*16+15
        //      of row r (contiguous, matching the B fragment).
        v16bf afrag;
        {
            const float* ap = xr + kt * 32 + h * 16;
            float4 a0 = *(const float4*)(ap + 0);
            float4 a1 = *(const float4*)(ap + 4);
            float4 a2 = *(const float4*)(ap + 8);
            float4 a3 = *(const float4*)(ap + 12);
            afrag[0]  = (__bf16)a0.x; afrag[1]  = (__bf16)a0.y;
            afrag[2]  = (__bf16)a0.z; afrag[3]  = (__bf16)a0.w;
            afrag[4]  = (__bf16)a1.x; afrag[5]  = (__bf16)a1.y;
            afrag[6]  = (__bf16)a1.z; afrag[7]  = (__bf16)a1.w;
            afrag[8]  = (__bf16)a2.x; afrag[9]  = (__bf16)a2.y;
            afrag[10] = (__bf16)a2.z; afrag[11] = (__bf16)a2.w;
            afrag[12] = (__bf16)a3.x; afrag[13] = (__bf16)a3.y;
            afrag[14] = (__bf16)a3.z; afrag[15] = (__bf16)a3.w;
        }
#pragma unroll
        for (int ct = 0; ct < 4; ++ct) {
            // ---- B fragment straight from LDS: lane (col=ct*16+r, half=h)
            //      holds K = h*16 .. h*16+15 -> 32 contiguous bytes.
            const v16bf bfrag =
                *(const v16bf*)(wlds + (ct * 16 + r) * D_DIM + kt * 32 + h * 16);

            acc[ct] = __builtin_amdgcn_wmma_f32_16x16x32_bf16(
                /*neg_a=*/false, afrag, /*neg_b=*/false, bfrag,
                /*c_mod=*/(short)0, acc[ct],
                /*reuse_a=*/false, /*reuse_b=*/false);
        }
    }

    // ---- Retire w from LDS, overlay logit buffer.
    __syncthreads();

    // ---- Spill logit tile to LDS (C layout: lane half h -> M = j + 8h, N = r)
#pragma unroll
    for (int ct = 0; ct < 4; ++ct)
#pragma unroll
        for (int j = 0; j < 8; ++j)
            slog[wid * 16 + j + h * 8][ct * 16 + r] = acc[ct][j];

    __syncthreads();

    // ---- Fused softmax / argmax: one thread per row (128 rows per block)
    if (threadIdx.x < 128) {
        const int t = threadIdx.x;
        float l0 = slog[t][0] + bg[0];
        float mx = l0;
        int   am = 0;
        float best = l0;
#pragma unroll 4
        for (int e = 1; e < E_DIM; ++e) {
            float l = slog[t][e] + bg[e];
            mx = fmaxf(mx, l);
            if (l > best) { best = l; am = e; }   // first-occurrence tie-break
        }
        float s = 0.f;
#pragma unroll 4
        for (int e = 0; e < E_DIM; ++e) {
            float l = slog[t][e] + bg[e];
            s += expf(l - mx);
        }
        const size_t gi = row0 + t;
        p0[gi]   = expf(l0 - mx) / s;
        amax[gi] = am;
    }
}

// ---------------------------------------------------------------------------
// Kernel 2: per batch b, member set over n in [0,64): member[n] = 1 iff some
// token's top-1 expert index equals n. masked0[b,n] = member ? p0 : 0.
// ---------------------------------------------------------------------------
__global__ __launch_bounds__(128)
void mask_kernel(const float* __restrict__ p0,
                 const int*   __restrict__ amax,
                 float* __restrict__ masked0)
{
    __shared__ int member[E_DIM];
    const int b = blockIdx.x;
    const int n = threadIdx.x;          // 0..127
    if (n < E_DIM) member[n] = 0;
    __syncthreads();
    const size_t gi = (size_t)b * N_SEQ + n;
    member[amax[gi]] = 1;               // benign same-value race
    __syncthreads();
    const float m = (n < E_DIM && member[n]) ? 1.0f : 0.0f;
    masked0[gi] = m * p0[gi];
}

// ---------------------------------------------------------------------------
// Kernel 3: per n, deterministic tree reduction over b of masked0 (float sum)
// and of (masked0 > 0) (load count). Grid = 128 blocks x 256 threads.
// ---------------------------------------------------------------------------
__global__ __launch_bounds__(256)
void denom_kernel(const float* __restrict__ masked0,
                  float* __restrict__ denomraw,
                  int*   __restrict__ loadcnt)
{
    __shared__ float sf[256];
    __shared__ int   si[256];
    const int n = blockIdx.x;
    const int t = threadIdx.x;
    float s = 0.f; int c = 0;
    for (int b = t; b < B_DIM; b += 256) {
        float v = masked0[(size_t)b * N_SEQ + n];
        s += v;
        c += (v > 0.f) ? 1 : 0;
    }
    sf[t] = s; si[t] = c;
    __syncthreads();
#pragma unroll
    for (int off = 128; off > 0; off >>= 1) {
        if (t < off) { sf[t] += sf[t + off]; si[t] += si[t + off]; }
        __syncthreads();
    }
    if (t == 0) { denomraw[n] = sf[0]; loadcnt[n] = si[0]; }
}

// ---------------------------------------------------------------------------
// Kernel 4: write full gs output [B,N,E]. One thread per token writes its 64
// floats (only e=0 may be nonzero). Grid = 1024 x 256.
// ---------------------------------------------------------------------------
__global__ __launch_bounds__(256)
void write_gs(const float* __restrict__ masked0,
              const float* __restrict__ denomraw,
              float* __restrict__ out)
{
    const size_t i = (size_t)blockIdx.x * 256 + threadIdx.x;  // token index
    const int n = (int)(i & (N_SEQ - 1));
    const float g = masked0[i] / (denomraw[n] + EPS_G) * CAP;
    float4* o = (float4*)(out + i * E_DIM);
    const float4 z = {0.f, 0.f, 0.f, 0.f};
    const float4 f = {g,   0.f, 0.f, 0.f};
    o[0] = f;
#pragma unroll
    for (int j = 1; j < 16; ++j) o[j] = z;
}

// ---------------------------------------------------------------------------
// Kernel 5: loss = cv^2(importance) + cv^2(load) over flattened [N,E]=8192
// values; only 128 (n, e=0) entries can be nonzero, rest are structural zeros.
// importance[n] = CAP * d/(d+eps); load[n] = count. One block, 128 threads.
// ---------------------------------------------------------------------------
__global__ __launch_bounds__(128)
void loss_kernel(const float* __restrict__ denomraw,
                 const int*   __restrict__ loadcnt,
                 float* __restrict__ loss_out)
{
    __shared__ float s1[128], s2[128], s3[128], s4[128];
    const int t = threadIdx.x;
    const float d = denomraw[t];
    const float v = CAP * d / (d + EPS_G);     // importance[n,0]
    const float l = (float)loadcnt[t];         // load[n,0]
    s1[t] = v; s2[t] = v * v; s3[t] = l; s4[t] = l * l;
    __syncthreads();
#pragma unroll
    for (int off = 64; off > 0; off >>= 1) {
        if (t < off) {
            s1[t] += s1[t + off]; s2[t] += s2[t + off];
            s3[t] += s3[t + off]; s4[t] += s4[t + off];
        }
        __syncthreads();
    }
    if (t == 0) {
        const float NT = (float)(N_SEQ * E_DIM);     // 8192 flattened elements
        float m1 = s1[0] / NT;
        float var1 = (s2[0] - NT * m1 * m1) / (NT - 1.0f);   // ddof=1
        float cv1 = var1 / (m1 * m1 + EPS_CV);
        float m2 = s3[0] / NT;
        float var2 = (s4[0] - NT * m2 * m2) / (NT - 1.0f);
        float cv2 = var2 / (m2 * m2 + EPS_CV);
        loss_out[0] = cv1 + cv2;
    }
}

// ---------------------------------------------------------------------------
extern "C" void kernel_launch(void* const* d_in, const int* in_sizes, int n_in,
                              void* d_out, int out_size, void* d_ws, size_t ws_size,
                              hipStream_t stream)
{
    (void)in_sizes; (void)n_in; (void)out_size; (void)ws_size;

    const float* x  = (const float*)d_in[0];   // [B,N,D] fp32
    const float* w  = (const float*)d_in[1];   // [E,D]   fp32
    const float* bg = (const float*)d_in[2];   // [E]     fp32
    float* out = (float*)d_out;                // gs [B,N,E] then loss scalar

    // Workspace layout (~3.2 MB)
    float*  p0       = (float*)d_ws;                 // [M_TOT]
    float*  masked0  = p0 + M_TOT;                   // [M_TOT]
    int*    amax     = (int*)(masked0 + M_TOT);      // [M_TOT]
    float*  denomraw = (float*)(amax + M_TOT);       // [N_SEQ]
    int*    loadcnt  = (int*)(denomraw + N_SEQ);     // [N_SEQ]
    __bf16* wbf      = (__bf16*)(loadcnt + N_SEQ);   // [E_DIM*D_DIM] bf16

    cvt_w_kernel     <<<(E_DIM * D_DIM) / 256, 256, 0, stream>>>(w, wbf);
    gate_gemm_softmax<<<M_TOT / 128, 256, 0, stream>>>(x, wbf, bg, p0, amax);
    mask_kernel      <<<B_DIM,       128, 0, stream>>>(p0, amax, masked0);
    denom_kernel     <<<N_SEQ,       256, 0, stream>>>(masked0, denomraw, loadcnt);
    write_gs         <<<M_TOT / 256, 256, 0, stream>>>(masked0, denomraw, out);
    loss_kernel      <<<1,           128, 0, stream>>>(denomraw, loadcnt,
                                                       out + (size_t)M_TOT * E_DIM);
}
